// TransformerBlock_29738353558136
// MI455X (gfx1250) — compile-verified, matrix operand order checked by reading, not run on hardware
//
#include <hip/hip_runtime.h>
#include <hip/hip_bf16.h>

#define B_  32
#define S_  512
#define H_  128
#define HEADS_ 4
#define DH_ 32
#define FF_ 512
#define N_  16
#define SN_ 64

#define USE_ASYNC_LDS 1

typedef __attribute__((ext_vector_type(16))) _Float16 v16h;
typedef __attribute__((ext_vector_type(8)))  float    v8f;

// ---------------------------------------------------------------------------
// WMMA fragment loaders (CDNA5 16x16x32 f16 layout, wave32).
// A (16x32): lane l holds row M=l&15; element e -> K = 8*(l>>4) + (e<8?e:e+8)
// B (32x16): lane l holds col N=l&15; same K mapping.
// C/D (16x16 f32): lane l, vgpr r -> (M = r + 8*(l>>4), N = l&15)
// ---------------------------------------------------------------------------
__device__ inline v16h frag_a_f32(const float* As, int lda, int mbase, int lane) {
  int m  = mbase + (lane & 15);
  int kb = (lane >> 4) << 3;
  const float* row = As + m * lda;
  v16h a;
#pragma unroll
  for (int e = 0; e < 16; ++e) {
    int k = kb + (e < 8 ? e : e + 8);
    a[e] = (_Float16)row[k];
  }
  return a;
}

__device__ inline v16h frag_b_f32(const float* Bs, int nbase, int ldn, int lane) {
  int n  = nbase + (lane & 15);
  int kb = (lane >> 4) << 3;
  v16h b;
#pragma unroll
  for (int e = 0; e < 16; ++e) {
    int k = kb + (e < 8 ? e : e + 8);
    b[e] = (_Float16)Bs[k * ldn + n];
  }
  return b;
}

__device__ inline v16h frag_b_f16(const _Float16* Bs, int nbase, int ldn, int lane) {
  int n  = nbase + (lane & 15);
  int kb = (lane >> 4) << 3;
  v16h b;
#pragma unroll
  for (int e = 0; e < 16; ++e) {
    int k = kb + (e < 8 ? e : e + 8);
    b[e] = Bs[k * ldn + n];
  }
  return b;
}

#if USE_ASYNC_LDS
// Per-lane async copy of 16B global -> LDS (ASYNCcnt-tracked).
__device__ inline void async_ld_b128(unsigned lds_off, const float* gaddr) {
  asm volatile("global_load_async_to_lds_b128 %0, %1, off"
               :: "v"(lds_off), "v"((unsigned long long)(uintptr_t)gaddr)
               : "memory");
}
__device__ inline void async_wait0() {
  asm volatile("s_wait_asynccnt 0x0" ::: "memory");
}
__device__ inline unsigned lds_off_of(const float* p) {
  return (unsigned)(uintptr_t)p;   // LDS aperture: addr[31:0] is the LDS offset
}
#endif

// ---------------------------------------------------------------------------
// WMMA GEMM: C[M,N] = epi(A[M,K] @ W[K,N]).  Block 256 thr = 8 waves.
// Block tile 64x64, BK=64; each wave computes a 16x32 C strip (2 acc, 4 wmma
// per staging round).  Tiles staged f32 in LDS via async global->LDS copies.
// EPI: 0 none, 1 +bias, 2 +bias+residual, 3 gelu(+bias)
// ---------------------------------------------------------------------------
#define LDA_ 68   // 64 + 4 pad floats: keeps 16B alignment, breaks bank aliasing

template <int EPI>
__global__ __launch_bounds__(256) void gemm_kernel(const float* __restrict__ A,
                                                   const float* __restrict__ W,
                                                   const float* __restrict__ bias,
                                                   const float* __restrict__ resid,
                                                   float* __restrict__ C,
                                                   int M, int Nd, int Kd) {
  __shared__ float As[64 * LDA_];
  __shared__ float Bs[64 * 64];

  const int tid  = threadIdx.x;
  const int lane = tid & 31;
  const int wv   = tid >> 5;
  const int bm   = blockIdx.x * 64;
  const int bn   = blockIdx.y * 64;
  const int tr   = wv >> 1;      // 0..3
  const int tc   = wv & 1;       // 0..1

  v8f acc0 = {};
  v8f acc1 = {};

  for (int k0 = 0; k0 < Kd; k0 += 64) {
#if USE_ASYNC_LDS
    {
      const int rhalf = lane >> 4;          // 0/1
      const int chunk = lane & 15;          // 16B chunk id
#pragma unroll
      for (int i = 0; i < 4; ++i) {
        int row = wv * 8 + i * 2 + rhalf;   // 0..63
        int col = chunk * 4;                // f32 col, 16B aligned
        async_ld_b128(lds_off_of(&As[row * LDA_ + col]),
                      A + (size_t)(bm + row) * Kd + k0 + col);
        async_ld_b128(lds_off_of(&Bs[row * 64 + col]),
                      W + (size_t)(k0 + row) * Nd + bn + col);
      }
    }
    if (k0 + 64 < Kd) {      // prefetch next K panel into L2
      int row = tid >> 2, cc = (tid & 3) * 16;
      __builtin_prefetch(A + (size_t)(bm + row) * Kd + k0 + 64 + cc, 0, 1);
      __builtin_prefetch(W + (size_t)(k0 + 64 + row) * Nd + bn + cc, 0, 1);
    }
    async_wait0();
    __syncthreads();
#else
    {
      int m  = tid >> 2;
      int kc = (tid & 3) << 4;
      const float* srcA = A + (size_t)(bm + m) * Kd + k0 + kc;
      const float* srcB = W + (size_t)(k0 + m) * Nd + bn + kc;
#pragma unroll
      for (int j = 0; j < 16; ++j) {
        As[m * LDA_ + kc + j] = srcA[j];
        Bs[m * 64   + kc + j] = srcB[j];
      }
    }
    __syncthreads();
#endif

#pragma unroll
    for (int kk = 0; kk < 64; kk += 32) {
      v16h a  = frag_a_f32(As + kk, LDA_, tr * 16, lane);
      v16h b0 = frag_b_f32(Bs + kk * 64, tc * 32,      64, lane);
      v16h b1 = frag_b_f32(Bs + kk * 64, tc * 32 + 16, 64, lane);
      acc0 = __builtin_amdgcn_wmma_f32_16x16x32_f16(false, a, false, b0,
                                                    (short)0, acc0, false, false);
      acc1 = __builtin_amdgcn_wmma_f32_16x16x32_f16(false, a, false, b1,
                                                    (short)0, acc1, false, false);
    }
    __syncthreads();
  }

  const int row0 = bm + tr * 16 + ((lane >> 4) << 3);
  const int col0 = bn + tc * 32 + (lane & 15);
  const int col1 = col0 + 16;
  float bv0 = (EPI >= 1) ? bias[col0] : 0.f;
  float bv1 = (EPI >= 1) ? bias[col1] : 0.f;
#pragma unroll
  for (int r = 0; r < 8; ++r) {
    float t0 = acc0[r] + bv0;
    float t1 = acc1[r] + bv1;
    if (EPI == 3) {
      float a3 = t0 * t0 * t0;
      t0 = 0.5f * t0 * (1.f + tanhf(0.7978845608028654f * (t0 + 0.044715f * a3)));
      float b3 = t1 * t1 * t1;
      t1 = 0.5f * t1 * (1.f + tanhf(0.7978845608028654f * (t1 + 0.044715f * b3)));
    }
    if (EPI == 2) {
      t0 += resid[(size_t)(row0 + r) * Nd + col0];
      t1 += resid[(size_t)(row0 + r) * Nd + col1];
    }
    C[(size_t)(row0 + r) * Nd + col0] = t0;
    C[(size_t)(row0 + r) * Nd + col1] = t1;
  }
}

// ---------------------------------------------------------------------------
// LayerNorm over H=128, one block (128 thr) per row.
// ---------------------------------------------------------------------------
__global__ __launch_bounds__(128) void ln_kernel(const float* __restrict__ X,
                                                 const float* __restrict__ g,
                                                 const float* __restrict__ bb,
                                                 float* __restrict__ Y) {
  const int r = blockIdx.x, t = threadIdx.x;
  __shared__ float red[128];
  float v = X[(size_t)r * H_ + t];
  red[t] = v;
  __syncthreads();
  for (int s = 64; s > 0; s >>= 1) { if (t < s) red[t] += red[t + s]; __syncthreads(); }
  float mean = red[0] * (1.f / H_);
  __syncthreads();
  float d = v - mean;
  red[t] = d * d;
  __syncthreads();
  for (int s = 64; s > 0; s >>= 1) { if (t < s) red[t] += red[t + s]; __syncthreads(); }
  float var = red[0] * (1.f / H_);
  Y[(size_t)r * H_ + t] = d * rsqrtf(var + 1e-6f) * g[t] + bb[t];
}

// ---------------------------------------------------------------------------
// LDS-tiled online-softmax attention.
// Block = 8 waves sharing (b,h); wave w owns query q = bx*8+w.
// Per 32-key tile: K/V staged cooperatively (coalesced, stride-33 padded);
// scores with lane=key (q broadcast from LDS); P.V with lane=dim (p broadcast).
// ---------------------------------------------------------------------------
__global__ __launch_bounds__(256) void attn_kernel(const float* __restrict__ Q,
                                                   const float* __restrict__ Km,
                                                   const float* __restrict__ Vm,
                                                   const int* __restrict__ mask,
                                                   float* __restrict__ O) {
  __shared__ float Ks[32 * 33];
  __shared__ float Vs[32 * 33];
  __shared__ float qsh[8 * 32];
  __shared__ float psh[8 * 32];

  const int tid  = threadIdx.x;
  const int lane = tid & 31;
  const int wv   = tid >> 5;
  const int q    = blockIdx.x * 8 + wv;
  const int h    = blockIdx.y;
  const int b    = blockIdx.z;

  const size_t qbase = ((size_t)b * S_ + q) * H_ + h * DH_;
  qsh[wv * 32 + lane] = Q[qbase + lane];
  const int* mrow = mask + ((size_t)b * S_ + q) * S_;

  float m = -3e30f, l = 0.f, acc = 0.f;

  const int key  = tid >> 3;        // 0..31  (staging role)
  const int c4   = (tid & 7) * 4;   // 0,4,..28

  for (int kt = 0; kt < 16; ++kt) {
    // stage K/V tile (32 keys x 32 dims), float4 coalesced
    {
      size_t rbase = ((size_t)b * S_ + kt * 32 + key) * H_ + h * DH_ + c4;
      float4 k4 = *(const float4*)(Km + rbase);
      float4 v4 = *(const float4*)(Vm + rbase);
      Ks[key * 33 + c4 + 0] = k4.x; Ks[key * 33 + c4 + 1] = k4.y;
      Ks[key * 33 + c4 + 2] = k4.z; Ks[key * 33 + c4 + 3] = k4.w;
      Vs[key * 33 + c4 + 0] = v4.x; Vs[key * 33 + c4 + 1] = v4.y;
      Vs[key * 33 + c4 + 2] = v4.z; Vs[key * 33 + c4 + 3] = v4.w;
      if (kt + 1 < 16) {
        __builtin_prefetch(Km + rbase + 32 * H_, 0, 1);
        __builtin_prefetch(Vm + rbase + 32 * H_, 0, 1);
      }
    }
    __syncthreads();

    // scores: lane = key index within tile
    float s = 0.f;
#pragma unroll
    for (int d = 0; d < DH_; ++d) s += qsh[wv * 32 + d] * Ks[lane * 33 + d];
    s *= 0.1767766952966369f;                 // 1/sqrt(32)
    if (mrow[kt * 32 + lane] == 0) s = -1e9f;

    float tmx = s;
#pragma unroll
    for (int off = 16; off > 0; off >>= 1) tmx = fmaxf(tmx, __shfl_xor(tmx, off));
    float mn   = fmaxf(m, tmx);
    float corr = __expf(m - mn);
    float p    = __expf(s - mn);
    float ts   = p;
#pragma unroll
    for (int off = 16; off > 0; off >>= 1) ts += __shfl_xor(ts, off);
    l = l * corr + ts;
    psh[wv * 32 + lane] = p;

    // P.V: lane = dim
    float vacc = 0.f;
#pragma unroll
    for (int k2 = 0; k2 < 32; ++k2) vacc += psh[wv * 32 + k2] * Vs[k2 * 33 + lane];
    acc = acc * corr + vacc;
    m = mn;
    __syncthreads();
  }
  O[qbase + lane] = acc / l;
}

// ---------------------------------------------------------------------------
// Neighbor branch fused per (n,b): only last-token output is needed, so only
// q(last) is formed; K/V matrices never materialized (hoisted sums).
// ---------------------------------------------------------------------------
__global__ __launch_bounds__(256) void neigh_kernel(const float* __restrict__ nbs,
                                                    const int* __restrict__ nmask,
                                                    const float* __restrict__ g1,
                                                    const float* __restrict__ b1,
                                                    const float* __restrict__ wq,
                                                    const float* __restrict__ bq,
                                                    const float* __restrict__ wk,
                                                    const float* __restrict__ bk,
                                                    const float* __restrict__ wvm,
                                                    const float* __restrict__ bv,
                                                    const float* __restrict__ wo,
                                                    const float* __restrict__ bo,
                                                    float* __restrict__ XNEI) {
  __shared__ float xl[SN_ * H_];
  __shared__ float qlast[H_];
  __shared__ float u[HEADS_ * H_];
  __shared__ float kb0[HEADS_];
  __shared__ float prob[HEADS_ * SN_];
  __shared__ float tac[HEADS_ * H_];
  __shared__ float olast[H_];

  const int n = blockIdx.x, b = blockIdx.y;
  const int tid = threadIdx.x, lane = tid & 31, wv = tid >> 5;
  const size_t base = (size_t)(n * B_ + b) * SN_ * H_;

  for (int r = wv; r < SN_; r += 8) {
    float vals[4]; float s = 0.f;
#pragma unroll
    for (int j = 0; j < 4; ++j) { int c = lane + 32 * j; vals[j] = nbs[base + (size_t)r * H_ + c]; s += vals[j]; }
#pragma unroll
    for (int off = 16; off > 0; off >>= 1) s += __shfl_xor(s, off);
    float mean = s * (1.f / H_);
    float sq = 0.f;
#pragma unroll
    for (int j = 0; j < 4; ++j) { float d = vals[j] - mean; sq += d * d; }
#pragma unroll
    for (int off = 16; off > 0; off >>= 1) sq += __shfl_xor(sq, off);
    float rs = rsqrtf(sq * (1.f / H_) + 1e-6f);
#pragma unroll
    for (int j = 0; j < 4; ++j) { int c = lane + 32 * j; xl[r * H_ + c] = (vals[j] - mean) * rs * g1[c] + b1[c]; }
  }
  __syncthreads();

  if (tid < H_) {
    float a = bq[tid];
    for (int c = 0; c < H_; ++c) a += xl[(SN_ - 1) * H_ + c] * wq[c * H_ + tid];
    qlast[tid] = a;
  }
  __syncthreads();

  for (int idx = tid; idx < HEADS_ * H_; idx += 256) {
    int h = idx >> 7, c = idx & 127;
    float a = 0.f;
    for (int d = 0; d < DH_; ++d) a += qlast[h * DH_ + d] * wk[c * H_ + h * DH_ + d];
    u[idx] = a;
  }
  if (tid < HEADS_) {
    float a = 0.f;
    for (int d = 0; d < DH_; ++d) a += qlast[tid * DH_ + d] * bk[tid * DH_ + d];
    kb0[tid] = a;
  }
  __syncthreads();

  {
    int h = tid >> 6, s = tid & 63;
    float sc = kb0[h];
    for (int c = 0; c < H_; ++c) sc += xl[s * H_ + c] * u[h * H_ + c];
    sc *= 0.1767766952966369f;
    size_t mi = ((size_t)(n * B_ + b) * SN_ + (SN_ - 1)) * SN_ + s;
    if (nmask[mi] == 0) sc = -1e9f;
    prob[h * SN_ + s] = sc;
  }
  __syncthreads();

  if (tid < HEADS_) {
    float mx = -1e30f;
    for (int s = 0; s < SN_; ++s) mx = fmaxf(mx, prob[tid * SN_ + s]);
    float sm = 0.f;
    for (int s = 0; s < SN_; ++s) { float e = __expf(prob[tid * SN_ + s] - mx); prob[tid * SN_ + s] = e; sm += e; }
    float inv = 1.f / sm;
    for (int s = 0; s < SN_; ++s) prob[tid * SN_ + s] *= inv;
  }
  __syncthreads();

  for (int idx = tid; idx < HEADS_ * H_; idx += 256) {
    int h = idx >> 7, c = idx & 127;
    float a = 0.f;
    for (int s = 0; s < SN_; ++s) a += prob[h * SN_ + s] * xl[s * H_ + c];
    tac[idx] = a;
  }
  __syncthreads();

  if (tid < H_) {
    int h = tid >> 5;
    float a = bv[tid];
    for (int c = 0; c < H_; ++c) a += tac[h * H_ + c] * wvm[c * H_ + tid];
    olast[tid] = a;
  }
  __syncthreads();

  if (tid < H_) {
    float a = bo[tid];
    for (int c = 0; c < H_; ++c) a += olast[c] * wo[c * H_ + tid];
    XNEI[((size_t)b * N_ + n) * H_ + tid] = nbs[base + (size_t)(SN_ - 1) * H_ + tid] + a;
  }
}

// ---------------------------------------------------------------------------
// Prep: Wp = W1a - W1c, Wsum = W1c + W1d   (au_w1 is (4H, H) row-major)
// ---------------------------------------------------------------------------
__global__ void prep_au(const float* __restrict__ au_w1,
                        float* __restrict__ Wp, float* __restrict__ Wsum) {
  int i = blockIdx.x * 256 + threadIdx.x;
  int k = i >> 7, j = i & 127;
  float a  = au_w1[k * H_ + j];
  float c  = au_w1[(2 * H_ + k) * H_ + j];
  float dd = au_w1[(3 * H_ + k) * H_ + j];
  Wp[i]   = a - c;
  Wsum[i] = c + dd;
}

// ---------------------------------------------------------------------------
// Fused merge-attention: cross term (x2*ne)@W1b via WMMA with the product
// A-fragment built on the fly; epilogue adds P+R, relu, reduces with au_w2;
// then softmax over n and final x2 + att@x_nei.
// ---------------------------------------------------------------------------
__global__ __launch_bounds__(256) void au_kernel(const float* __restrict__ X2,
                                                 const float* __restrict__ XNEI,
                                                 const float* __restrict__ P,
                                                 const float* __restrict__ R,
                                                 const float* __restrict__ W1b,
                                                 const float* __restrict__ au_w2,
                                                 const float* __restrict__ au_b2,
                                                 float* __restrict__ Out) {
  __shared__ float    x2s[16 * H_];
  __shared__ float    nes[N_ * H_];
  __shared__ _Float16 w1bs[H_ * H_];
  __shared__ float    att[16 * N_];
  __shared__ float    w2s[H_];

  const int s0 = blockIdx.x * 16, b = blockIdx.y;
  const int tid = threadIdx.x, lane = tid & 31, wv = tid >> 5;

  for (int i = tid; i < 16 * H_; i += 256) {
    x2s[i] = X2[((size_t)b * S_ + s0 + (i >> 7)) * H_ + (i & 127)];
    nes[i] = XNEI[((size_t)b * N_ + (i >> 7)) * H_ + (i & 127)];
  }
  for (int i = tid; i < H_ * H_; i += 256) w1bs[i] = (_Float16)W1b[i];
  if (tid < H_) w2s[tid] = au_w2[tid];
  __syncthreads();

  const float bias2 = au_b2[0];
  const size_t pbase = ((size_t)b * S_ + s0) * H_;

  for (int t2 = 0; t2 < 2; ++t2) {
    const int nsel = wv + t2 * 8;
    float psum[8];
#pragma unroll
    for (int r = 0; r < 8; ++r) psum[r] = 0.f;

    for (int nt = 0; nt < 8; ++nt) {
      v8f acc = {};
#pragma unroll
      for (int k0 = 0; k0 < H_; k0 += 32) {
        const int mm = lane & 15;
        const int kb = (lane >> 4) << 3;
        const float* xr = x2s + mm * H_ + k0;
        const float* nr = nes + nsel * H_ + k0;
        v16h a;
#pragma unroll
        for (int e = 0; e < 16; ++e) {
          int k = kb + (e < 8 ? e : e + 8);
          a[e] = (_Float16)(xr[k] * nr[k]);
        }
        v16h bf = frag_b_f16(w1bs + k0 * H_, nt * 16, H_, lane);
        acc = __builtin_amdgcn_wmma_f32_16x16x32_f16(false, a, false, bf,
                                                     (short)0, acc, false, false);
      }
      const int col = nt * 16 + (lane & 15);
      const float rv  = R[((size_t)b * N_ + nsel) * H_ + col];
      const float w2v = w2s[col];
      const int row0  = (lane >> 4) << 3;
#pragma unroll
      for (int r = 0; r < 8; ++r) {
        float t = acc[r] + P[pbase + (size_t)(row0 + r) * H_ + col] + rv;
        t = fmaxf(t, 0.f);
        psum[r] += t * w2v;
      }
    }
#pragma unroll
    for (int r = 0; r < 8; ++r)
      for (int off = 8; off > 0; off >>= 1) psum[r] += __shfl_xor(psum[r], off);
    if ((lane & 15) == 0) {
      int rbase = (lane >> 4) << 3;
#pragma unroll
      for (int r = 0; r < 8; ++r) att[(rbase + r) * N_ + nsel] = psum[r] + bias2;
    }
  }
  __syncthreads();

  if (tid < 16) {
    float mx = -1e30f;
    for (int n = 0; n < N_; ++n) mx = fmaxf(mx, att[tid * N_ + n]);
    float sm = 0.f;
    for (int n = 0; n < N_; ++n) { float e = __expf(att[tid * N_ + n] - mx); att[tid * N_ + n] = e; sm += e; }
    float inv = 1.f / sm;
    for (int n = 0; n < N_; ++n) att[tid * N_ + n] *= inv;
  }
  __syncthreads();

  for (int i = tid; i < 16 * H_; i += 256) {
    int row = i >> 7, hh = i & 127;
    float o = x2s[i];
    for (int n = 0; n < N_; ++n) o += att[row * N_ + n] * nes[n * H_ + hh];
    Out[((size_t)b * S_ + s0 + row) * H_ + hh] = o;
  }
}

// ---------------------------------------------------------------------------
extern "C" void kernel_launch(void* const* d_in, const int* in_sizes, int n_in,
                              void* d_out, int out_size, void* d_ws, size_t ws_size,
                              hipStream_t stream) {
  const float* x      = (const float*)d_in[0];
  const int*   mask   = (const int*)  d_in[1];
  const float* neighs = (const float*)d_in[2];
  const int*   nmask  = (const int*)  d_in[3];
  const float* wq = (const float*)d_in[4];  const float* bq = (const float*)d_in[5];
  const float* wk = (const float*)d_in[6];  const float* bk = (const float*)d_in[7];
  const float* wvv= (const float*)d_in[8];  const float* bv = (const float*)d_in[9];
  const float* wo = (const float*)d_in[10]; const float* bo = (const float*)d_in[11];
  const float* ln1g = (const float*)d_in[12]; const float* ln1b = (const float*)d_in[13];
  const float* ln2g = (const float*)d_in[14]; const float* ln2b = (const float*)d_in[15];
  const float* ffw1 = (const float*)d_in[16]; const float* ffb1 = (const float*)d_in[17];
  const float* ffw2 = (const float*)d_in[18]; const float* ffb2 = (const float*)d_in[19];
  const float* auw1 = (const float*)d_in[20]; const float* aub1 = (const float*)d_in[21];
  const float* auw2 = (const float*)d_in[22]; const float* aub2 = (const float*)d_in[23];
  float* out = (float*)d_out;
  float* ws  = (float*)d_ws;

  const int M = B_ * S_;                 // 16384
  // workspace layout (floats), with region reuse:
  const size_t A   = 0;                  // 2M : LN1 out, later LN2 out
  const size_t Qo  = 2097152;            // 2M
  const size_t Ko  = 4194304;            // 2M
  const size_t Vo  = 6291456;            // 2M
  const size_t AOo = 8388608;            // 2M
  const size_t X1o = 10485760;           // 2M
  const size_t FFHo= 2097152;            // 8M : reuses Q/K/V/AO after attention
  const size_t X2o = 0;                  // 2M : reuses A after FFN gemm1
  const size_t Po  = 10485760;           // 2M : reuses X1 after FFN gemm2
  const size_t XNEIo = 12582912;         // 64K
  const size_t Ro    = 12648448;         // 64K
  const size_t Wpo   = 12713984;         // 16K
  const size_t Wso   = 12730368;         // 16K

  // 1) LN1
  ln_kernel<<<M, 128, 0, stream>>>(x, ln1g, ln1b, ws + A);
  // 2) Q/K/V projections (WMMA)
  gemm_kernel<1><<<dim3(M/64, 2), 256, 0, stream>>>(ws + A, wq, bq, nullptr, ws + Qo, M, H_, H_);
  gemm_kernel<1><<<dim3(M/64, 2), 256, 0, stream>>>(ws + A, wk, bk, nullptr, ws + Ko, M, H_, H_);
  gemm_kernel<1><<<dim3(M/64, 2), 256, 0, stream>>>(ws + A, wvv, bv, nullptr, ws + Vo, M, H_, H_);
  // 3) attention (LDS-tiled online softmax)
  attn_kernel<<<dim3(S_/8, HEADS_, B_), 256, 0, stream>>>(ws + Qo, ws + Ko, ws + Vo, mask, ws + AOo);
  // 4) output proj + residual
  gemm_kernel<2><<<dim3(M/64, 2), 256, 0, stream>>>(ws + AOo, wo, bo, x, ws + X1o, M, H_, H_);
  // 5) neighbor branch (fused)
  neigh_kernel<<<dim3(N_, B_), 256, 0, stream>>>(neighs, nmask, ln1g, ln1b,
                                                 wq, bq, wk, bk, wvv, bv, wo, bo, ws + XNEIo);
  // 6) LN2
  ln_kernel<<<M, 128, 0, stream>>>(ws + X1o, ln2g, ln2b, ws + A);
  // 7) FFN gemm1 + gelu
  gemm_kernel<3><<<dim3(M/64, FF_/64), 256, 0, stream>>>(ws + A, ffw1, ffb1, nullptr, ws + FFHo, M, FF_, H_);
  // 8) FFN gemm2 + bias + residual
  gemm_kernel<2><<<dim3(M/64, 2), 256, 0, stream>>>(ws + FFHo, ffw2, ffb2, ws + X1o, ws + X2o, M, H_, FF_);
  // 9) merge-layer weight decomposition
  prep_au<<<64, 256, 0, stream>>>(auw1, ws + Wpo, ws + Wso);
  // 10) P = X2 @ (W1a - W1c)
  gemm_kernel<0><<<dim3(M/64, 2), 256, 0, stream>>>(ws + X2o, ws + Wpo, nullptr, nullptr, ws + Po, M, H_, H_);
  // 11) R = XNEI @ (W1c + W1d) + au_b1
  gemm_kernel<1><<<dim3((B_*N_)/64, 2), 256, 0, stream>>>(ws + XNEIo, ws + Wso, aub1, nullptr, ws + Ro, B_*N_, H_, H_);
  // 12) fused cross-term WMMA + relu + au_w2 + softmax + combine
  au_kernel<<<dim3(S_/16, B_), 256, 0, stream>>>(ws + X2o, ws + XNEIo, ws + Po, ws + Ro,
                                                 auw1 + H_ * H_, auw2, aub2, out);
}
